// ROIAlignSpatialAttention_87780541596077
// MI455X (gfx1250) — matrix operand order repaired and checked by reading, not compiled, not run on hardware
//
#include <hip/hip_runtime.h>

typedef __attribute__((ext_vector_type(16))) _Float16 v16h;
typedef __attribute__((ext_vector_type(8)))  float    v8f;

#define Bc     8
#define Nn     4096
#define Hh     11
#define Wdim   176
#define HW     (Hh * Wdim)      // 1936 = 121 * 16 exactly
#define CIN    72
#define COUT   18
#define PTILES (HW / 16)        // 121
#define CONV_WAVES (Bc * PTILES) // 968 = 121 blocks * 8 waves
#define CE_THREADS (Bc * Nn * 25) // 819200 = 3200 * 256
#define CE_BLOCKS  (CE_THREADS / 256)

// ---------------------------------------------------------------------------
// Kernel 1: 1x1 conv 72->18 as WMMA GEMM.  out18 layout: [B, HW, 18] (channel-
// contiguous per pixel so the gather stage reads 18 contiguous floats).
// One wave per (batch, 16-pixel tile); 2 M-tiles x 3 K-steps = 6 WMMAs/wave.
// ---------------------------------------------------------------------------
__global__ __launch_bounds__(256)
void conv_wmma_kernel(const float* __restrict__ feat,   // [B,72,H,W]
                      const float* __restrict__ Wc,     // [18,72]
                      const float* __restrict__ bc,     // [18]
                      float* __restrict__ out18)        // [B,HW,18]
{
    const int wid   = blockIdx.x * (blockDim.x >> 5) + (threadIdx.x >> 5);
    const int lane  = threadIdx.x & 31;
    const int b     = wid / PTILES;
    const int pt    = wid - b * PTILES;
    const int m     = lane & 15;        // A row / D column selector
    const int khalf = lane >> 4;        // 0: lanes 0-15, 1: lanes 16-31
    const int pcol  = pt * 16 + m;      // this lane's pixel column (B/D: N = lane&15)

    const float* fb = feat + (size_t)b * CIN * HW;

    v8f c0 = {0.f, 0.f, 0.f, 0.f, 0.f, 0.f, 0.f, 0.f};  // output rows 0..15
    v8f c1 = {0.f, 0.f, 0.f, 0.f, 0.f, 0.f, 0.f, 0.f};  // output rows 16..17 (padded)

    for (int ks = 0; ks < 3; ++ks) {
        const int cbase = ks * 32;
        v16h a0, a1, btile;
#pragma unroll
        for (int j = 0; j < 16; ++j) {
            // A (16-bit 16x32): lane = M; element j -> K
            const int ka = (j & 7) + 8 * khalf + 16 * (j >> 3);
            const int ca = cbase + ka;
            const float w0 = (ca < CIN)            ? Wc[m * CIN + ca]        : 0.f;
            const float w1 = (ca < CIN && m < 2)   ? Wc[(16 + m) * CIN + ca] : 0.f;
            a0[j] = (_Float16)w0;
            a1[j] = (_Float16)w1;
            // B (16-bit 32x16): lane&15 = N column; element j -> K = (j&7) + 8*khalf + 16*(j>>3)
            const int kb = (j & 7) + 8 * khalf + 16 * (j >> 3);
            const int cb = cbase + kb;
            const float fv = (cb < CIN) ? fb[(size_t)cb * HW + pcol] : 0.f;
            btile[j] = (_Float16)fv;
        }
        c0 = __builtin_amdgcn_wmma_f32_16x16x32_f16(false, a0, false, btile,
                                                    (short)0, c0, false, false);
        c1 = __builtin_amdgcn_wmma_f32_16x16x32_f16(false, a1, false, btile,
                                                    (short)0, c1, false, false);
    }

    // D layout: VGPR v -> M = v + 8*khalf, N = lane&15.
    float* op = out18 + ((size_t)b * HW + pcol) * COUT;
#pragma unroll
    for (int v = 0; v < 8; ++v) {
        const int o = v + 8 * khalf;
        op[o] = c0[v] + bc[o];
    }
    if (khalf == 0) {           // second M-tile: only rows 16,17 are real
        op[16] = c1[0] + bc[16];
        op[17] = c1[1] + bc[17];
    }
}

// ---------------------------------------------------------------------------
// Kernel 2: box -> 5x5 nearest grid-sample -> 6-way 4-class CE per point.
// One thread per (b, n, sample point). Per-block partial sums (deterministic).
// ---------------------------------------------------------------------------
__global__ __launch_bounds__(256)
void roi_ce_kernel(const float* __restrict__ rois,     // [B,N,7]
                   const float* __restrict__ labels,   // [B,6*HW]
                   const float* __restrict__ feat18,   // [B,HW,18]
                   float* __restrict__ partial)        // [CE_BLOCKS]
{
    const int t = blockIdx.x * blockDim.x + threadIdx.x;  // exact: < CE_THREADS
    const int b   = t / (Nn * 25);
    const int r   = t - b * (Nn * 25);
    const int n   = r / 25;
    const int p25 = r - n * 25;
    const int ai  = p25 / 5;         // width-sample index (gz axis)
    const int bi  = p25 - ai * 5;    // height-sample index (gx axis)

    const float* rp = rois + ((size_t)b * Nn + n) * 7;
    const float x0 = rp[0], z0 = rp[2];
    const float dx = rp[3], dy = rp[4], dz = rp[5], th = rp[6];

    const float half = (dy * cosf(th) + dx * sinf(th)) * 0.5f;
    const float x1 = x0 - half, x2 = x0 + half;
    const float z1 = z0 - dz * 0.5f, z2 = z0 + dz * 0.5f;

    const float gz  = z1 + (z2 - z1) * (ai * 0.25f);
    const float gx  = x1 + (x2 - x1) * (bi * 0.25f);
    const float gzn = (gz - 1.0f) * 0.25f;             // /4
    const float gxn = (gx - 35.2f) * (1.0f / 70.4f);

    const int iw = (int)rintf((gzn + 1.0f) * 0.5f * (float)(Wdim - 1));
    const int ih = (int)rintf((gxn + 1.0f) * 0.5f * (float)(Hh - 1));
    const bool valid = (iw >= 0) && (iw < Wdim) && (ih >= 0) && (ih < Hh);
    const int iwc = min(max(iw, 0), Wdim - 1);
    const int ihc = min(max(ih, 0), Hh - 1);
    const int idx = ihc * Wdim + iwc;
    const float vm = valid ? 1.f : 0.f;

    const float* fp = feat18 + ((size_t)b * HW + idx) * COUT;
    const float* lp = labels + (size_t)b * 6 * HW + idx;

    float lsum = 0.f;
#pragma unroll
    for (int g = 0; g < 6; ++g) {
        const float f0 = fp[3 * g + 0] * vm;
        const float f1 = fp[3 * g + 1] * vm;
        const float f2 = fp[3 * g + 2] * vm;
        float lab = lp[(size_t)g * HW] * vm;
        lab = (lab >= 0.f) ? lab : 0.f;                // cared mask (sg*(sg>=0))
        const int tgt = (int)lab;
        const float mx  = fmaxf(fmaxf(f0, f1), fmaxf(f2, 0.f));
        const float lse = mx + logf(expf(0.f - mx) + expf(f0 - mx) +
                                    expf(f1 - mx) + expf(f2 - mx));
        const float chosen = (tgt == 0) ? 0.f
                            : (tgt == 1) ? f0
                            : (tgt == 2) ? f1 : f2;
        lsum += lse - chosen;
    }

    __shared__ float sred[256];
    sred[threadIdx.x] = lsum;
    __syncthreads();
#pragma unroll
    for (int s = 128; s > 0; s >>= 1) {
        if (threadIdx.x < s) sred[threadIdx.x] += sred[threadIdx.x + s];
        __syncthreads();
    }
    if (threadIdx.x == 0) partial[blockIdx.x] = sred[0];
}

// ---------------------------------------------------------------------------
// Kernel 3: deterministic final reduction (double accum), scale by 1/M.
// ---------------------------------------------------------------------------
__global__ __launch_bounds__(256)
void final_reduce_kernel(const float* __restrict__ partial, float* __restrict__ out)
{
    __shared__ double sred[256];
    double s = 0.0;
    for (int i = threadIdx.x; i < CE_BLOCKS; i += 256) s += (double)partial[i];
    sred[threadIdx.x] = s;
    __syncthreads();
#pragma unroll
    for (int st = 128; st > 0; st >>= 1) {
        if (threadIdx.x < st) sred[threadIdx.x] += sred[threadIdx.x + st];
        __syncthreads();
    }
    if (threadIdx.x == 0)
        out[0] = (float)(sred[0] * (1.0 / (double)(Nn * 25 * 6)));
}

// ---------------------------------------------------------------------------
extern "C" void kernel_launch(void* const* d_in, const int* in_sizes, int n_in,
                              void* d_out, int out_size, void* d_ws, size_t ws_size,
                              hipStream_t stream)
{
    (void)in_sizes; (void)n_in; (void)out_size; (void)ws_size;
    const float* rois   = (const float*)d_in[0];  // [B,N,7]
    const float* feat   = (const float*)d_in[1];  // [B,72,H,W]
    const float* labels = (const float*)d_in[2];  // [B,6*H*W]
    const float* Wc     = (const float*)d_in[3];  // [18,72]
    const float* bc     = (const float*)d_in[4];  // [18]

    float* feat18  = (float*)d_ws;                         // B*HW*18 floats (~1.1 MB)
    float* partial = feat18 + (size_t)Bc * HW * COUT;      // CE_BLOCKS floats

    conv_wmma_kernel<<<CONV_WAVES / 8, 256, 0, stream>>>(feat, Wc, bc, feat18);
    roi_ce_kernel<<<CE_BLOCKS, 256, 0, stream>>>(rois, labels, feat18, partial);
    final_reduce_kernel<<<1, 256, 0, stream>>>(partial, (float*)d_out);
}
